// MyRNN_48524540510626
// MI455X (gfx1250) — compile-verified, matrix operand order checked by reading, not run on hardware
//
#include <hip/hip_runtime.h>
#include <cstdint>
#include <cstddef>

typedef __bf16 bf16;
typedef __attribute__((ext_vector_type(16))) __bf16 v16bf;
typedef __attribute__((ext_vector_type(8)))  __bf16 v8bf;
typedef __attribute__((ext_vector_type(4)))  __bf16 v4bf;
typedef __attribute__((ext_vector_type(2)))  __bf16 v2bf;
typedef __attribute__((ext_vector_type(8)))  float  v8f;

#define NHID   1024
#define NFEAT  1024
#define TSTEPS 128
#define BATCH  512

constexpr int BM = 128, BN = 128, BK = 32;
constexpr int LDA_STRIDE = 40;  // bf16/row: 80B -> 16B-aligned b128 reads, conflict-free banks
constexpr int LDB_STRIDE = 40;  // same for the column-major B tile

// ---------------------------------------------------------------------------
// Pack f32 weights [K][N] -> bf16, K-pair interleaved:
//   Wp[2*((k/2)*N + n)] = {w[k][n], w[k+1][n]}
// ---------------------------------------------------------------------------
__global__ __launch_bounds__(256)
void pack_weights(const float* __restrict__ w, bf16* __restrict__ wp, int K, int N) {
    int idx = blockIdx.x * blockDim.x + threadIdx.x;   // pair index
    int total = (K / 2) * N;
    if (idx >= total) return;
    int n  = idx % N;
    int kp = idx / N;
    float a = w[(size_t)(2 * kp)     * N + n];
    float b = w[(size_t)(2 * kp + 1) * N + n];
    v2bf p;
    p[0] = (bf16)a;
    p[1] = (bf16)b;
    *(v2bf*)&wp[2 * (size_t)idx] = p;
}

// ---------------------------------------------------------------------------
// WMMA GEMM, 128x128 block tile, BK=32, 8 wave32 per block, wave tile 32x64.
// 2-stage LDS double buffer (single staging regset -> spill-free), one
// barrier per K-step. All 12 fragment ds_load_b128 batched -> one dscnt
// wait, then 8 back-to-back v_wmma per iteration; global_prefetch_b8 two
// tiles ahead covers HBM latency, cross-wave bursts cover the rest.
// MODE 0: out_bf = bf16(A_f32 @ Bp + bias)                (xw precompute)
// MODE 1: s = tanh(A_bf16 @ Bp + add); out_bf = bf16(s);
//         if (outF) outF = s                              (recurrent step)
// ---------------------------------------------------------------------------
template <int MODE>
__global__ __launch_bounds__(256)
void rnn_gemm(const void* __restrict__ Aptr, const bf16* __restrict__ Bp,
              const float* __restrict__ bias,
              const bf16* __restrict__ addp, long addStride,
              bf16* __restrict__ outBf, float* __restrict__ outF,
              int M, int N, int K)
{
    __shared__ __align__(16) bf16 ldsA[2][BM * LDA_STRIDE];   // 2 x 10240 B
    __shared__ __align__(16) bf16 ldsB[2][BN * LDB_STRIDE];   // 2 x 10240 B

    const int tid  = threadIdx.x;
    const int lane = tid & 31;
    const int wave = tid >> 5;
    const int half = lane >> 4;      // 0: lanes 0-15, 1: lanes 16-31
    const int l16  = lane & 15;
    const int wm   = wave & 3;       // 4 waves along M (32 rows each)
    const int wn   = wave >> 2;      // 2 waves along N (64 cols each)

    const int rowBase = blockIdx.y * BM;
    const int colBase = blockIdx.x * BN;

    v8f acc[2][4];
#pragma unroll
    for (int i = 0; i < 2; ++i)
#pragma unroll
        for (int j = 0; j < 4; ++j)
            acc[i][j] = v8f{0.f, 0.f, 0.f, 0.f, 0.f, 0.f, 0.f, 0.f};

    // ---- per-thread staging geometry (hoisted out of the K loop) ----
    const int ar  = tid >> 3;               // A tile row 0..31 (+32 per pass)
    const int ac  = (tid & 7) * 4;          // A tile col 0..28
    const int bn0 = (tid & 63) * 2;         // B tile col pair base 0..126
    const int bkq = tid >> 6;               // B pair-row quadrant 0..3

    const float* aGf = (const float*)Aptr + (size_t)(rowBase + ar) * K + ac;
    const bf16*  aGh = (const bf16*) Aptr + (size_t)(rowBase + ar) * K + ac;
    const bf16*  bG  = Bp + 2 * ((size_t)bkq * N + colBase + bn0);

    const int aW = ar * LDA_STRIDE + ac;                     // LDS A store base
    const int bW = bn0 * LDB_STRIDE;                         // LDS B store base
    const int aR = (wm * 32 + l16) * LDA_STRIDE + 8 * half;  // LDS A frag base
    const int bR = (wn * 64 + l16) * LDB_STRIDE + 8 * half;  // LDS B frag base

    const int KSTEPS = K / BK;
    const size_t B_ADV = (size_t)32 * N;   // bf16 per K-step (16 pair-rows)

    v4bf aReg[4], bReg[4];                 // single staging set: spill-free

    auto loadTile = [&](int t) {
        const float* af = aGf + (size_t)t * BK;
        const bf16*  ah = aGh + (size_t)t * BK;
        const bf16*  bg = bG  + (size_t)t * B_ADV;
#pragma unroll
        for (int p = 0; p < 4; ++p) {
            if (MODE == 0) {
                const float4 f = *(const float4*)(af + (size_t)p * 32 * K);
                aReg[p][0] = (bf16)f.x; aReg[p][1] = (bf16)f.y;
                aReg[p][2] = (bf16)f.z; aReg[p][3] = (bf16)f.w;
            } else {
                aReg[p] = *(const v4bf*)(ah + (size_t)p * 32 * K);
            }
            bReg[p] = *(const v4bf*)(bg + (size_t)p * 8 * N);
        }
    };
    auto commitTile = [&](int b) {
#pragma unroll
        for (int p = 0; p < 4; ++p) {
            *(v4bf*)&ldsA[b][aW + p * 32 * LDA_STRIDE] = aReg[p];
            int kp = p * 4 + bkq;
            v2bf c0; c0[0] = bReg[p][0]; c0[1] = bReg[p][1];
            v2bf c1; c1[0] = bReg[p][2]; c1[1] = bReg[p][3];
            *(v2bf*)&ldsB[b][bW + 2 * kp]              = c0;
            *(v2bf*)&ldsB[b][bW + LDB_STRIDE + 2 * kp] = c1;
        }
    };

    // ---- stage tile 0 into buffer 0 ----
    loadTile(0);
    commitTile(0);
    __syncthreads();

    for (int it = 0; it < KSTEPS; ++it) {
        const int cur = it & 1;
        const int nxt = cur ^ 1;

        // ---- issue global loads for the NEXT tile ----
        if (it + 1 < KSTEPS) {
            loadTile(it + 1);
            if (it + 2 < KSTEPS) {   // global_prefetch two tiles ahead
                __builtin_prefetch(aGf + (size_t)(it + 2) * BK,    0, 1);
                __builtin_prefetch(bG  + (size_t)(it + 2) * B_ADV, 0, 1);
            }
        }

        // ---- all fragments batched: 12x ds_load_b128, one dscnt wait ----
        v16bf afrag[2], bfrag[4];
#pragma unroll
        for (int i = 0; i < 2; ++i) {
            v8bf lo = *(const v8bf*)&ldsA[cur][aR + i * 16 * LDA_STRIDE];
            v8bf hi = *(const v8bf*)&ldsA[cur][aR + i * 16 * LDA_STRIDE + 16];
            afrag[i] = __builtin_shufflevector(lo, hi, 0, 1, 2, 3, 4, 5, 6, 7,
                                               8, 9, 10, 11, 12, 13, 14, 15);
        }
#pragma unroll
        for (int j = 0; j < 4; ++j) {
            v8bf lo = *(const v8bf*)&ldsB[cur][bR + j * 16 * LDB_STRIDE];
            v8bf hi = *(const v8bf*)&ldsB[cur][bR + j * 16 * LDB_STRIDE + 16];
            bfrag[j] = __builtin_shufflevector(lo, hi, 0, 1, 2, 3, 4, 5, 6, 7,
                                               8, 9, 10, 11, 12, 13, 14, 15);
        }

        // ---- commit next tile to the other buffer ----
        if (it + 1 < KSTEPS) commitTile(nxt);

        // ---- 8x v_wmma_f32_16x16x32_bf16, back-to-back ----
#pragma unroll
        for (int i = 0; i < 2; ++i)
#pragma unroll
            for (int j = 0; j < 4; ++j)
                acc[i][j] = __builtin_amdgcn_wmma_f32_16x16x32_bf16(
                    false, afrag[i], false, bfrag[j],
                    (short)0, acc[i][j], false, false);

        __syncthreads();
    }

    // ---- epilogue: C/D layout — VGPR g holds row (g + 8*half), col = l16 ----
#pragma unroll
    for (int i = 0; i < 2; ++i) {
#pragma unroll
        for (int j = 0; j < 4; ++j) {
            int c = colBase + wn * 64 + j * 16 + l16;
            float bv = (MODE == 0) ? bias[c] : 0.0f;
#pragma unroll
            for (int g = 0; g < 8; ++g) {
                int r = rowBase + wm * 32 + i * 16 + half * 8 + g;
                float v = acc[i][j][g];
                if (MODE == 0) {
                    outBf[(size_t)r * N + c] = (bf16)(v + bv);
                } else {
                    float s = tanhf(v + (float)addp[(size_t)r * addStride + c]);
                    outBf[(size_t)r * N + c] = (bf16)s;
                    if (outF) outF[(size_t)r * N + c] = s;
                }
            }
        }
    }
}

// ---------------------------------------------------------------------------
extern "C" void kernel_launch(void* const* d_in, const int* in_sizes, int n_in,
                              void* d_out, int out_size, void* d_ws, size_t ws_size,
                              hipStream_t stream) {
    (void)in_sizes; (void)n_in; (void)out_size; (void)ws_size;

    const float* x  = (const float*)d_in[0];   // [512,128,1024]
    const float* wx = (const float*)d_in[1];   // [1024,1024]
    const float* wh = (const float*)d_in[2];   // [1024,1024]
    const float* bi = (const float*)d_in[3];   // [1,1024]
    float* out = (float*)d_out;                // [512,1024]

    // workspace layout (bf16 elements)
    bf16* wxp = (bf16*)d_ws;                          //  NFEAT*NHID        (2 MB)
    bf16* whp = wxp + (size_t)NFEAT * NHID;           //  NHID*NHID         (2 MB)
    bf16* hA  = whp + (size_t)NHID * NHID;            //  BATCH*NHID        (1 MB)
    bf16* hB  = hA  + (size_t)BATCH * NHID;           //  BATCH*NHID        (1 MB)
    bf16* xw  = hB  + (size_t)BATCH * NHID;           //  BATCH*TSTEPS*NHID (128 MB)

    // pack both weight matrices to bf16, K-pair interleaved
    {
        int total = (NFEAT / 2) * NHID;
        int blocks = (total + 255) / 256;
        pack_weights<<<blocks, 256, 0, stream>>>(wx, wxp, NFEAT, NHID);
        pack_weights<<<blocks, 256, 0, stream>>>(wh, whp, NHID, NHID);
    }

    // h0 = 0
    hipMemsetAsync(hA, 0, (size_t)BATCH * NHID * sizeof(bf16), stream);

    // xw[b*T + t][h] = x @ wx + bi   (M = 65536, N = K = 1024)
    {
        dim3 grid(NHID / BN, (BATCH * TSTEPS) / BM);   // (8, 512)
        rnn_gemm<0><<<grid, 256, 0, stream>>>(x, wxp, bi, nullptr, 0,
                                              xw, nullptr,
                                              BATCH * TSTEPS, NHID, NFEAT);
    }

    // 128 dependent steps: h = tanh(xw_t + h @ wh)
    {
        dim3 grid(NHID / BN, BATCH / BM);              // (8, 4)
        bf16* hp = hA;
        bf16* hn = hB;
        for (int t = 0; t < TSTEPS; ++t) {
            rnn_gemm<1><<<grid, 256, 0, stream>>>(
                hp, whp, nullptr,
                xw + (size_t)t * NHID, (long)TSTEPS * NHID,
                hn, (t == TSTEPS - 1) ? out : nullptr,
                BATCH, NHID, NHID);
            bf16* tmp = hp; hp = hn; hn = tmp;
        }
    }
}
